// ServerPrototypeAggregator_9740985828022
// MI455X (gfx1250) — matrix operand order repaired and checked by reading, not compiled, not run on hardware
//
#include <hip/hip_runtime.h>
#include <hip/hip_bf16.h>

// ---------------------------------------------------------------------------
// Fused per-class transformer encoder layer + cross-attention for MI455X
// (gfx1250, wave32, WMMA f32_16x16x32_f16).
//
// dims: M=64 clients, C=2000 classes, D=512, H=4 heads, HD=128, DFF=2048.
// One block (8 wave32s) per class; all intermediates live in LDS (278 KB
// dynamic, within the 320KB/WGP budget); all GEMMs via v_wmma_f32_16x16x32_f16
// with f16 weights pre-converted into d_ws (L2-resident, 192MB L2 >> 6MB).
// ---------------------------------------------------------------------------

#define M_ 64
#define C_ 2000
#define D_ 512
#define H_ 4
#define HD_ 128
#define DFF_ 2048
#define LDX 520            // padded leading dim for 64x512 f16 buffers
#define LDQ 136            // padded leading dim for 64x128 f16 buffers
#define LDT 72             // padded leading dim for 128x64 (V^T) / 64x64 (P)
#define ATTN_SCALE 0.08838834764831845f  // 1/sqrt(128)

typedef _Float16 v16h __attribute__((ext_vector_type(16)));
typedef _Float16 v8h  __attribute__((ext_vector_type(8)));
typedef float    v8f  __attribute__((ext_vector_type(8)));

__device__ __forceinline__ v8f zero8() {
  v8f z = {0.f, 0.f, 0.f, 0.f, 0.f, 0.f, 0.f, 0.f};
  return z;
}

// A fragment (16x32 f16, row-major source, leading dim `ld` in halves).
// Per ISA: lanes 0-15 hold K 0..7 & 16..23 of row m; lanes 16-31 hold
// K 8..15 & 24..31.  Two contiguous 16B chunks per lane.
__device__ __forceinline__ v16h load_a_frag(const _Float16* base, int ld,
                                            int m0, int k0, int lane) {
  const int m  = m0 + (lane & 15);
  const int kb = k0 + ((lane >> 4) << 3);
  const _Float16* p = base + m * ld + kb;
  v8h lo = *(const v8h*)p;
  v8h hi = *(const v8h*)(p + 16);
  v16h a;
#pragma unroll
  for (int i = 0; i < 8; ++i) { a[i] = lo[i]; a[i + 8] = hi[i]; }
  return a;
}

// B fragment (32x16 f16). Source is row-major (N, K) => column-major B, so
// lane n reads K like A: lanes 0-15 K 0..7 & 16..23, lanes 16-31 K 8..15 & 24..31.
__device__ __forceinline__ v16h load_b_frag(const _Float16* base, int ld,
                                            int n0, int k0, int lane) {
  const int n  = n0 + (lane & 15);
  const int kb = k0 + ((lane >> 4) << 3);
  const _Float16* p = base + (size_t)n * ld + kb;
  v8h lo = *(const v8h*)p;
  v8h hi = *(const v8h*)(p + 16);
  v16h b;
#pragma unroll
  for (int i = 0; i < 8; ++i) { b[i] = lo[i]; b[i + 8] = hi[i]; }
  return b;
}

__device__ __forceinline__ v8f wmma_f16(v16h a, v16h b, v8f c) {
  return __builtin_amdgcn_wmma_f32_16x16x32_f16(false, a, false, b,
                                                (short)0, c, false, false);
}

// In-place row LayerNorm on a 64xD_ f16 LDS buffer (ld = LDX), 8 rows/wave.
__device__ __forceinline__ void ln_rows_inplace(_Float16* buf, const float* g,
                                                const float* b, int wave,
                                                int lane) {
#pragma unroll 1
  for (int r = 0; r < 8; ++r) {
    const int m = wave * 8 + r;
    float xs[16];
    float s = 0.f, sq = 0.f;
#pragma unroll
    for (int i = 0; i < 16; ++i) {
      float v = (float)buf[m * LDX + lane + 32 * i];
      xs[i] = v; s += v; sq += v * v;
    }
#pragma unroll
    for (int o = 16; o; o >>= 1) {
      s  += __shfl_xor(s, o, 32);
      sq += __shfl_xor(sq, o, 32);
    }
    const float mu  = s * (1.f / (float)D_);
    const float var = sq * (1.f / (float)D_) - mu * mu;
    const float inv = rsqrtf(var + 1e-5f);
#pragma unroll
    for (int i = 0; i < 16; ++i) {
      const int d = lane + 32 * i;
      buf[m * LDX + d] = (_Float16)((xs[i] - mu) * inv * g[d] + b[d]);
    }
  }
}

// fp32 -> f16 weight conversion (weights shared by all classes; stay in L2)
__global__ __launch_bounds__(256) void cvt_f16_kernel(
    const float* __restrict__ src, _Float16* __restrict__ dst, int n) {
  int i = blockIdx.x * blockDim.x + threadIdx.x;
  if (i < n) dst[i] = (_Float16)src[i];
}

__global__ __launch_bounds__(256) void ServerPrototypeAggregator_fused_kernel(
    const float* __restrict__ protos,      // (M, C, D)
    const float* __restrict__ g0, const float* __restrict__ b0,
    const _Float16* __restrict__ Wqkv,     // (3D, D) f16
    const float* __restrict__ bqkv,        // (3D,)
    const _Float16* __restrict__ Wo,       // (D, D) f16
    const float* __restrict__ bo,          // (D,)
    const _Float16* __restrict__ W1,       // (DFF, D) f16
    const float* __restrict__ b1,          // (DFF,)
    const _Float16* __restrict__ W2,       // (D, DFF) f16
    const float* __restrict__ b2,          // (D,)
    const float* __restrict__ g1, const float* __restrict__ b1n,
    const float* __restrict__ g2, const float* __restrict__ b2n,
    const float* __restrict__ gq,          // (2000, D) global_query
    const int*   __restrict__ labels,      // (C,)
    float*       __restrict__ out)         // (C, D)
{
  extern __shared__ char smem[];
  _Float16* Xh  = (_Float16*)smem;           // 64 x LDX : x (post first LN)
  _Float16* Yh  = Xh  + M_ * LDX;            // 64 x LDX : self-attn output SA
  _Float16* Zh  = Yh  + M_ * LDX;            // 64 x LDX : running activations
  _Float16* Qs  = Zh  + M_ * LDX;            // 64 x LDQ : Q (head), pre-scaled
  _Float16* Ks  = Qs  + M_ * LDQ;            // 64 x LDQ : K (head)
  _Float16* VsT = Ks  + M_ * LDQ;            // 128 x LDT : V^T (head)
  _Float16* Ps  = VsT + HD_ * LDT;           // 64 x LDT : softmax probs f16
  float*    Sf  = (float*)(Ps + M_ * LDT);   // 64 x 64 f32 scores / scratch

  const int c    = blockIdx.x;
  const int tid  = threadIdx.x;
  const int lane = tid & 31;
  const int wave = tid >> 5;

  // ---- Stage 1: load protos[:, c, :], LayerNorm(norm_g, norm_b) -> Xh f16
#pragma unroll 1
  for (int r = 0; r < 8; ++r) {
    const int m = wave * 8 + r;
    const float* row = protos + (size_t)m * (C_ * D_) + (size_t)c * D_;
    float xs[16];
    float s = 0.f, sq = 0.f;
#pragma unroll
    for (int i = 0; i < 16; ++i) {
      float v = row[lane + 32 * i];
      xs[i] = v; s += v; sq += v * v;
    }
#pragma unroll
    for (int o = 16; o; o >>= 1) {
      s  += __shfl_xor(s, o, 32);
      sq += __shfl_xor(sq, o, 32);
    }
    const float mu  = s * (1.f / (float)D_);
    const float var = sq * (1.f / (float)D_) - mu * mu;
    const float inv = rsqrtf(var + 1e-5f);
#pragma unroll
    for (int i = 0; i < 16; ++i) {
      const int d = lane + 32 * i;
      Xh[m * LDX + d] = (_Float16)((xs[i] - mu) * inv * g0[d] + b0[d]);
    }
  }
  __syncthreads();

  // ---- Stage 2: multi-head self-attention, head by head, SA -> Yh
#pragma unroll 1
  for (int h = 0; h < H_; ++h) {
    // Q/K/V for this head: 3 x (64x128) = 96 16x16 tiles, K = 512
#pragma unroll 1
    for (int t = wave; t < 96; t += 8) {
      const int mat = t >> 5;            // 0=Q 1=K 2=V
      const int tt  = t & 31;
      const int m0  = (tt & 3) * 16;
      const int n0  = (tt >> 2) * 16;
      const int wrow = mat * D_ + h * HD_ + n0;   // row in Wqkv (3D x D)
      v8f acc = zero8();
#pragma unroll
      for (int kk = 0; kk < D_; kk += 32) {
        v16h a = load_a_frag(Xh, LDX, m0, kk, lane);
        v16h b = load_b_frag(Wqkv, D_, wrow, kk, lane);
        acc = wmma_f16(a, b, acc);
      }
      const int n  = n0 + (lane & 15);
      const int mb = m0 + ((lane >> 4) << 3);
      const float bias = bqkv[mat * D_ + h * HD_ + n];
      if (mat == 0) {
#pragma unroll
        for (int r = 0; r < 8; ++r)
          Qs[(mb + r) * LDQ + n] = (_Float16)((acc[r] + bias) * ATTN_SCALE);
      } else if (mat == 1) {
#pragma unroll
        for (int r = 0; r < 8; ++r)
          Ks[(mb + r) * LDQ + n] = (_Float16)(acc[r] + bias);
      } else {  // V stored transposed so P@V's B operand is K-contiguous
#pragma unroll
        for (int r = 0; r < 8; ++r)
          VsT[n * LDT + (mb + r)] = (_Float16)(acc[r] + bias);
      }
    }
    __syncthreads();

    // S = Q K^T : 64x64, K = 128 (scale already folded into Q)
#pragma unroll 1
    for (int t = wave; t < 16; t += 8) {
      const int m0 = (t & 3) * 16;
      const int n0 = (t >> 2) * 16;
      v8f acc = zero8();
#pragma unroll
      for (int kk = 0; kk < HD_; kk += 32) {
        v16h a = load_a_frag(Qs, LDQ, m0, kk, lane);
        v16h b = load_b_frag(Ks, LDQ, n0, kk, lane);   // K row n == B col n
        acc = wmma_f16(a, b, acc);
      }
      const int n  = n0 + (lane & 15);
      const int mb = m0 + ((lane >> 4) << 3);
#pragma unroll
      for (int r = 0; r < 8; ++r) Sf[(mb + r) * 64 + n] = acc[r];
    }
    __syncthreads();

    // row softmax -> Ps f16
#pragma unroll 1
    for (int r = 0; r < 8; ++r) {
      const int row = wave * 8 + r;
      float v0 = Sf[row * 64 + lane];
      float v1 = Sf[row * 64 + 32 + lane];
      float mx = fmaxf(v0, v1);
#pragma unroll
      for (int o = 16; o; o >>= 1) mx = fmaxf(mx, __shfl_xor(mx, o, 32));
      float e0 = __expf(v0 - mx), e1 = __expf(v1 - mx);
      float sm = e0 + e1;
#pragma unroll
      for (int o = 16; o; o >>= 1) sm += __shfl_xor(sm, o, 32);
      const float inv = 1.f / sm;
      Ps[row * LDT + lane]      = (_Float16)(e0 * inv);
      Ps[row * LDT + 32 + lane] = (_Float16)(e1 * inv);
    }
    __syncthreads();

    // O_h = P @ V : (64x64)@(64x128), K = 64; write into Yh cols [h*128, ...)
#pragma unroll 1
    for (int t = wave; t < 32; t += 8) {
      const int m0 = (t & 3) * 16;
      const int n0 = (t >> 2) * 16;
      v8f acc = zero8();
#pragma unroll
      for (int kk = 0; kk < M_; kk += 32) {
        v16h a = load_a_frag(Ps, LDT, m0, kk, lane);
        v16h b = load_b_frag(VsT, LDT, n0, kk, lane);  // V^T row n == B col n
        acc = wmma_f16(a, b, acc);
      }
      const int n  = h * HD_ + n0 + (lane & 15);
      const int mb = m0 + ((lane >> 4) << 3);
#pragma unroll
      for (int r = 0; r < 8; ++r) Yh[(mb + r) * LDX + n] = (_Float16)acc[r];
    }
    __syncthreads();
  }

  // ---- Stage 3: out_proj + residual + LN1 -> Zh
#pragma unroll 1
  for (int t = wave; t < 128; t += 8) {
    const int m0 = (t & 3) * 16;
    const int n0 = (t >> 2) * 16;
    v8f acc = zero8();
#pragma unroll
    for (int kk = 0; kk < D_; kk += 32) {
      v16h a = load_a_frag(Yh, LDX, m0, kk, lane);
      v16h b = load_b_frag(Wo, D_, n0, kk, lane);
      acc = wmma_f16(a, b, acc);
    }
    const int n  = n0 + (lane & 15);
    const int mb = m0 + ((lane >> 4) << 3);
    const float bias = bo[n];
#pragma unroll
    for (int r = 0; r < 8; ++r) {
      const int m = mb + r;
      Zh[m * LDX + n] =
          (_Float16)(acc[r] + bias + (float)Xh[m * LDX + n]);
    }
  }
  __syncthreads();
  ln_rows_inplace(Zh, g1, b1n, wave, lane);
  __syncthreads();

  // ---- Stage 4: FF = LN2(Zh + relu(Zh@W1^T + b1)@W2^T + b2), K chunked x4.
  // Per-wave 16 f32 accumulator tiles (128 VGPRs); hidden chunk reuses Xh.
  v8f facc[16];
#pragma unroll
  for (int i = 0; i < 16; ++i) facc[i] = zero8();
  const int fmt = wave & 3;            // accumulator tile-row
  const int fnb = (wave >> 2) * 16;    // accumulator tile-col base

#pragma unroll 1
  for (int j = 0; j < 4; ++j) {
    // H_j = relu(Zh @ W1[j*512 : (j+1)*512, :]^T + b1_j) -> Xh (reused)
#pragma unroll 1
    for (int t = wave; t < 128; t += 8) {
      const int m0 = (t & 3) * 16;
      const int n0 = (t >> 2) * 16;
      v8f acc = zero8();
#pragma unroll
      for (int kk = 0; kk < D_; kk += 32) {
        v16h a = load_a_frag(Zh, LDX, m0, kk, lane);
        v16h b = load_b_frag(W1, D_, j * D_ + n0, kk, lane);
        acc = wmma_f16(a, b, acc);
      }
      const int n  = n0 + (lane & 15);
      const int mb = m0 + ((lane >> 4) << 3);
      const float bias = b1[j * D_ + n];
#pragma unroll
      for (int r = 0; r < 8; ++r) {
        float v = acc[r] + bias;
        Xh[(mb + r) * LDX + n] = (_Float16)(v > 0.f ? v : 0.f);
      }
    }
    __syncthreads();

    // facc += H_j @ W2[:, j*512:(j+1)*512]^T   (A tile shared across 16 B's)
#pragma unroll 1
    for (int kk = 0; kk < D_; kk += 32) {
      v16h a = load_a_frag(Xh, LDX, fmt * 16, kk, lane);
#pragma unroll
      for (int i = 0; i < 16; ++i) {
        v16h b = load_b_frag(W2, DFF_, (fnb + i) * 16, j * D_ + kk, lane);
        facc[i] = wmma_f16(a, b, facc[i]);
      }
    }
    __syncthreads();
  }

  // finalize: Zh = Zh + facc + b2 (disjoint (m,n) per wave -> race-free)
  {
    const int n_lo = (lane & 15);
    const int mb   = fmt * 16 + ((lane >> 4) << 3);
#pragma unroll
    for (int i = 0; i < 16; ++i) {
      const int n = (fnb + i) * 16 + n_lo;
      const float bias = b2[n];
#pragma unroll
      for (int r = 0; r < 8; ++r) {
        const int m = mb + r;
        Zh[m * LDX + n] =
            (_Float16)(facc[i][r] + bias + (float)Zh[m * LDX + n]);
      }
    }
  }
  __syncthreads();
  ln_rows_inplace(Zh, g2, b2n, wave, lane);
  __syncthreads();

  // ---- Stage 5: cross-attention with global query (tiny: VALU)
  const int label = labels[c];
  const float* qv = gq + (size_t)label * D_;
  {
    const int h = tid >> 6;           // 0..3
    const int m = tid & 63;
    float acc = 0.f;
#pragma unroll 4
    for (int d = 0; d < HD_; ++d)
      acc += qv[h * HD_ + d] * (float)Zh[m * LDX + h * HD_ + d];
    Sf[h * 64 + m] = acc * ATTN_SCALE;
  }
  __syncthreads();
  if (wave < H_) {                    // one wave per head: softmax over M=64
    const int h = wave;
    float v0 = Sf[h * 64 + lane];
    float v1 = Sf[h * 64 + 32 + lane];
    float mx = fmaxf(v0, v1);
#pragma unroll
    for (int o = 16; o; o >>= 1) mx = fmaxf(mx, __shfl_xor(mx, o, 32));
    float e0 = __expf(v0 - mx), e1 = __expf(v1 - mx);
    float sm = e0 + e1;
#pragma unroll
    for (int o = 16; o; o >>= 1) sm += __shfl_xor(sm, o, 32);
    const float inv = 1.f / sm;
    Sf[h * 64 + lane]      = e0 * inv;
    Sf[h * 64 + 32 + lane] = e1 * inv;
  }
  __syncthreads();
#pragma unroll 1
  for (int d = tid; d < D_; d += 256) {
    const int h = d >> 7;
    float acc = 0.f;
#pragma unroll 4
    for (int m = 0; m < M_; ++m)
      acc += Sf[h * 64 + m] * (float)Zh[m * LDX + d];
    out[(size_t)c * D_ + d] = acc;
  }
}

extern "C" void kernel_launch(void* const* d_in, const int* in_sizes, int n_in,
                              void* d_out, int out_size, void* d_ws,
                              size_t ws_size, hipStream_t stream) {
  const float* protos    = (const float*)d_in[0];
  const float* norm_g    = (const float*)d_in[1];
  const float* norm_b    = (const float*)d_in[2];
  const float* in_proj_w = (const float*)d_in[3];
  const float* in_proj_b = (const float*)d_in[4];
  const float* out_prj_w = (const float*)d_in[5];
  const float* out_prj_b = (const float*)d_in[6];
  const float* lin1_w    = (const float*)d_in[7];
  const float* lin1_b    = (const float*)d_in[8];
  const float* lin2_w    = (const float*)d_in[9];
  const float* lin2_b    = (const float*)d_in[10];
  const float* norm1_g   = (const float*)d_in[11];
  const float* norm1_b   = (const float*)d_in[12];
  const float* norm2_g   = (const float*)d_in[13];
  const float* norm2_b   = (const float*)d_in[14];
  const float* gquery    = (const float*)d_in[15];
  const int*   labels    = (const int*)d_in[16];
  float*       out       = (float*)d_out;

  // f16 weight staging in workspace (6.3 MB; resident in 192MB L2)
  _Float16* ws = (_Float16*)d_ws;
  const int n_qkv = 3 * D_ * D_;      // 786432
  const int n_o   = D_ * D_;          // 262144
  const int n_1   = DFF_ * D_;        // 1048576
  const int n_2   = D_ * DFF_;        // 1048576
  _Float16* Wqkv = ws;
  _Float16* Wo   = Wqkv + n_qkv;
  _Float16* W1   = Wo + n_o;
  _Float16* W2   = W1 + n_1;

  cvt_f16_kernel<<<(n_qkv + 255) / 256, 256, 0, stream>>>(in_proj_w, Wqkv, n_qkv);
  cvt_f16_kernel<<<(n_o   + 255) / 256, 256, 0, stream>>>(out_prj_w, Wo, n_o);
  cvt_f16_kernel<<<(n_1   + 255) / 256, 256, 0, stream>>>(lin1_w,    W1, n_1);
  cvt_f16_kernel<<<(n_2   + 255) / 256, 256, 0, stream>>>(lin2_w,    W2, n_2);

  // dynamic LDS: 3*64*520 + 2*64*136 + 128*72 + 64*72 halves + 64*64 floats
  const size_t smem_bytes =
      (size_t)(3 * M_ * LDX + 2 * M_ * LDQ + HD_ * LDT + M_ * LDT) * 2 +
      (size_t)(M_ * 64) * 4;   // = 278528 B (<= 320KB WGP LDS)

  ServerPrototypeAggregator_fused_kernel<<<C_, 256, smem_bytes, stream>>>(
      protos, norm_g, norm_b, Wqkv, in_proj_b, Wo, out_prj_b, W1, lin1_b, W2,
      lin2_b, norm1_g, norm1_b, norm2_g, norm2_b, gquery, labels, out);
}
